// RNN_34282428957328
// MI455X (gfx1250) — compile-verified, matrix operand order checked by reading, not run on hardware
//
#include <hip/hip_runtime.h>
#include <hip/hip_bf16.h>

// ---------------- problem constants ----------------
#define H 1024
#define B 16
#define S 2000
#define O 8
#define NWG 16          // workgroups in persistent grid (cols per wg = 64)
#define TPB 128         // 4 waves of 32
#define DT 0.001f
#define TAU 0.01f
#define TAU_DEPR 0.2f
#define TAU_FACIL 1.5f
#define BETA 50.0f
#define ACT_GAIN 8.0f
#define ACT_THRESH 0.5f
#define NOISE_TAU 0.01f

typedef __attribute__((ext_vector_type(16))) __bf16 bf16x16;
typedef __attribute__((ext_vector_type(8)))  float  v8f;

static __device__ __forceinline__ unsigned short f2bf(float f) {
    unsigned u = __float_as_uint(f);
    unsigned r = (u + 0x7FFFu + ((u >> 16) & 1u)) >> 16;   // RNE
    return (unsigned short)r;
}
static __device__ __forceinline__ float sigmoidf(float x) {
    return 1.0f / (1.0f + __expf(-x));
}

// ---------------------------------------------------------------------------
// Prep: build bf16 B-fragments of Wm = W_hh * mask, laid out per-lane
// for v_wmma_f32_16x16x32_bf16; also zero the grid barrier counter.
// wfrag layout (halfs): [tile(64)][kb(32)][lane(32)][elem(16)]
// ---------------------------------------------------------------------------
__global__ void rnn_prep_kernel(const float* __restrict__ W_hh,
                                const float* __restrict__ W_hh_mask,
                                unsigned short* __restrict__ wfrag,
                                unsigned int* __restrict__ bar) {
    size_t gid = (size_t)blockIdx.x * 256 + threadIdx.x;
    if (gid == 0) *bar = 0u;
    if (gid >= (size_t)H * H) return;
    int tile = (int)(gid >> 14);          // 32*32*16 = 16384 per tile
    int rem  = (int)(gid & 16383);
    int kb   = rem >> 9;
    int rem2 = rem & 511;
    int lane = rem2 >> 4;
    int e    = rem2 & 15;
    int ncol = tile * 16 + (lane & 15);   // output column (row of Wm)
    int hh   = lane >> 4;
    int ko   = (e & 7) + 8 * hh + 16 * (e >> 3);
    int k    = kb * 32 + ko;
    float w = W_hh[(size_t)ncol * H + k] * W_hh_mask[(size_t)ncol * H + k];
    wfrag[gid] = f2bf(w);
}

// ---------------------------------------------------------------------------
// Persistent recurrent kernel: 16 workgroups x 128 threads, 2000 steps.
// Wave w of workgroup g owns columns [g*64 + w*16, +16) for all 16 batches,
// with state held in the WMMA C/D layout (v8f per state variable).
// ---------------------------------------------------------------------------
__global__ __launch_bounds__(TPB) void rnn_main_kernel(
    const float* __restrict__ I,      // [B,S,1]
    const float* __restrict__ h0,     // [B,H]
    const float* __restrict__ r0,
    const float* __restrict__ u0,
    const float* __restrict__ W_ih,   // [H,1]
    const float* __restrict__ off_ih, // [H]
    const float* __restrict__ p_rel,  // [H]
    float* __restrict__ out_n, float* __restrict__ out_h,
    float* __restrict__ out_r, float* __restrict__ out_u,
    const unsigned short* __restrict__ wfrag,   // [H/16][32][32][16] bf16
    unsigned short* __restrict__ sfrag,         // 2 x [32][32][16] bf16 (double buf)
    unsigned int* __restrict__ bar) {
    extern __shared__ unsigned short wlds[];    // 4 tiles * 16384 halfs = 128 KB

    const int wg   = blockIdx.x;
    const int tid  = threadIdx.x;
    const int wave = tid >> 5;
    const int lane = tid & 31;

    // ---- stage this workgroup's weight slice (128 KB) into LDS, once ----
    {
        const uint4* src = (const uint4*)(wfrag + (size_t)wg * 4 * 16384);
        uint4* dst = (uint4*)wlds;
        for (int i = tid; i < 8192; i += TPB) dst[i] = src[i];
    }
    __syncthreads();

    const int hh  = lane >> 4;
    const int col = wg * 64 + wave * 16 + (lane & 15);

    const float p      = p_rel[col];
    const float presyn = 1.0f / p;
    const float wih    = W_ih[col];
    const float oih    = off_ih[col];

    // scatter coordinates of this lane's column inside the A-fragment buffer
    const int kb_c    = col >> 5;
    const int ko      = col & 31;
    const int laneAdj = ((ko & 15) >= 8) ? 16 : 0;
    const int elem    = (ko & 7) + ((ko >= 16) ? 8 : 0);

    float h[8], r[8], u[8], ht[8], n[8];
#pragma unroll
    for (int i = 0; i < 8; ++i) {
        int m = i + 8 * hh;
        h[i]  = h0[(size_t)m * H + col];
        r[i]  = r0[(size_t)m * H + col];
        u[i]  = u0[(size_t)m * H + col];
        n[i]  = 0.0f;
        ht[i] = sigmoidf(ACT_GAIN * (h[i] - ACT_THRESH));
    }

    const unsigned short* wtile = wlds + (size_t)wave * 16384;

    for (int t = 0; t < S; ++t) {
        unsigned short* sb = sfrag + (size_t)(t & 1) * (32 * 32 * 16);

        // ---- phase 1: s = ht*r*u*presyn (old values), scatter as A-frag; r/u update ----
#pragma unroll
        for (int i = 0; i < 8; ++i) {
            int m = i + 8 * hh;
            float s = ht[i] * r[i] * u[i] * presyn;
            sb[((kb_c * 32 + m + laneAdj) << 4) + elem] = f2bf(s);
            float ro = r[i], uo = u[i];
            r[i] = ro + ((1.0f - ro) * (DT / TAU_DEPR) - (BETA * DT) * uo * ro * ht[i]);
            u[i] = uo + ((p - uo) * (DT / TAU_FACIL) + (BETA * DT) * p * (1.0f - uo) * ht[i]);
            size_t ob = (size_t)m * ((size_t)S * H) + (size_t)t * H + col;
            out_r[ob] = r[i];
            out_u[ob] = u[i];
        }

        // ---- device-scope barrier (monotone counter, one arrival per workgroup) ----
        __builtin_amdgcn_fence(__ATOMIC_RELEASE, "agent");
        __syncthreads();
        if (tid == 0) {
            __hip_atomic_fetch_add(bar, 1u, __ATOMIC_RELEASE, __HIP_MEMORY_SCOPE_AGENT);
            unsigned int target = (unsigned int)(t + 1) * NWG;
            while (__hip_atomic_load(bar, __ATOMIC_ACQUIRE, __HIP_MEMORY_SCOPE_AGENT) < target)
                __builtin_amdgcn_s_sleep(2);
        }
        __syncthreads();
        __builtin_amdgcn_fence(__ATOMIC_ACQUIRE, "agent");

        // ---- phase 2: rec[:, tile] = s @ Wm.T[:, tile] via 32 bf16 WMMAs ----
        v8f acc = {};
#pragma unroll 4
        for (int kb = 0; kb < 32; ++kb) {
            bf16x16 a = *(const bf16x16*)(sb + (((kb * 32) + lane) << 4));
            bf16x16 b = *(const bf16x16*)(wtile + (((kb * 32) + lane) << 4));
            acc = __builtin_amdgcn_wmma_f32_16x16x32_bf16(
                false, a, false, b, (short)0, acc, false, false);
        }

        // ---- state update: h, ht (rec arrives already in C/D layout) ----
#pragma unroll
        for (int i = 0; i < 8; ++i) {
            int m = i + 8 * hh;
            float It  = I[(size_t)m * S + t];
            float ext = It * wih + oih + n[i];     // uses OLD n (carry)
            n[i] *= (1.0f - DT / NOISE_TAU);
            h[i] += (-h[i] + acc[i] + ext) * (DT / TAU);
            ht[i] = sigmoidf(ACT_GAIN * (h[i] - ACT_THRESH));
            size_t ob = (size_t)m * ((size_t)S * H) + (size_t)t * H + col;
            out_n[ob] = n[i];
            out_h[ob] = h[i];
        }
    }
}

// ---------------------------------------------------------------------------
// Readout z = transfer(h) @ (W_hz*blockmask).T + offset_hz — no recurrence,
// fully parallel over (b,t). One wave per (b,t); block-diagonal mask applied
// implicitly (output o reads hidden block [o*128, (o+1)*128)).
// ---------------------------------------------------------------------------
__global__ __launch_bounds__(256) void rnn_z_kernel(
    const float* __restrict__ h_all,   // [B,S,H]
    const float* __restrict__ W_hz,    // [O,H]
    const float* __restrict__ off_hz,  // [O]
    float* __restrict__ z_out) {       // [B,S,O]
    int widx = (blockIdx.x * 256 + threadIdx.x) >> 5;   // wave id = (b,t)
    int lane = threadIdx.x & 31;
    if (widx >= B * S) return;
    const float* hp = h_all + (size_t)widx * H;
    float z[8];
#pragma unroll
    for (int o = 0; o < 8; ++o) z[o] = 0.0f;
#pragma unroll 4
    for (int j = 0; j < 32; ++j) {
        int k = lane + 32 * j;
        float t = sigmoidf(ACT_GAIN * (hp[k] - ACT_THRESH));
        int o = k >> 7;
        z[o] += t * W_hz[(size_t)o * H + k];
    }
#pragma unroll
    for (int off = 16; off >= 1; off >>= 1)
#pragma unroll
        for (int o = 0; o < 8; ++o) z[o] += __shfl_xor(z[o], off, 32);
    if (lane == 0) {
#pragma unroll
        for (int o = 0; o < 8; ++o) z_out[(size_t)widx * O + o] = z[o] + off_hz[o];
    }
}

// ---------------------------------------------------------------------------
extern "C" void kernel_launch(void* const* d_in, const int* in_sizes, int n_in,
                              void* d_out, int out_size, void* d_ws, size_t ws_size,
                              hipStream_t stream) {
    const float* I         = (const float*)d_in[0];
    const float* h0        = (const float*)d_in[1];
    const float* r0        = (const float*)d_in[2];
    const float* u0        = (const float*)d_in[3];
    const float* W_ih      = (const float*)d_in[4];
    const float* off_ih    = (const float*)d_in[5];
    const float* W_hh      = (const float*)d_in[6];
    const float* W_hz      = (const float*)d_in[7];
    const float* off_hz    = (const float*)d_in[8];
    const float* p_rel     = (const float*)d_in[9];
    const float* W_hh_mask = (const float*)d_in[10];
    // d_in[11] = W_hz_mask: block-diagonal, applied implicitly in rnn_z_kernel

    float* out   = (float*)d_out;
    const size_t NPB = (size_t)B * S * H;           // 32,768,000
    float* out_n = out;
    float* out_h = out + NPB;
    float* out_r = out + 2 * NPB;
    float* out_u = out + 3 * NPB;
    float* out_z = out + 4 * NPB;

    unsigned int*   bar   = (unsigned int*)d_ws;
    unsigned short* sfrag = (unsigned short*)((char*)d_ws + 256);            // 64 KB
    unsigned short* wfrag = (unsigned short*)((char*)d_ws + 256 + 65536);    // 2 MB

    // 1) build bf16 weight fragments + zero barrier
    rnn_prep_kernel<<<(H * H) / 256, 256, 0, stream>>>(W_hh, W_hh_mask, wfrag, bar);

    // 2) persistent recurrent kernel: 16 WGs x 128 threads, 128 KB LDS each
    (void)hipFuncSetAttribute((const void*)rnn_main_kernel,
                              hipFuncAttributeMaxDynamicSharedMemorySize, 131072);
    rnn_main_kernel<<<NWG, TPB, 131072, stream>>>(
        I, h0, r0, u0, W_ih, off_ih, p_rel,
        out_n, out_h, out_r, out_u, wfrag, sfrag, bar);

    // 3) readout over all steps (one wave per (b,t))
    rnn_z_kernel<<<(B * S) / 8, 256, 0, stream>>>(out_h, W_hz, off_hz, out_z);
}
